// PGN_2267742732892
// MI455X (gfx1250) — matrix operand order repaired and checked by reading, not compiled, not run on hardware
//
#include <hip/hip_runtime.h>
#include <hip/hip_bf16.h>

// Dims fixed by the reference: B=4, N=128, F=H=128, TRI=8.
constexpr int Bn = 4, Nn = 128, Fn = 128, Hn = 128;
constexpr int LDW = 136;                // padded bf16 row stride (272B) -> conflict-free ds_load_b128

typedef __attribute__((ext_vector_type(16))) __bf16 bf16x16;
typedef __attribute__((ext_vector_type(8)))  __bf16 bf16x8;
typedef __attribute__((ext_vector_type(8)))  float  f32x8;

// A-fragment 16x32 bf16 (rows in lanes 0..15 / 16..31 hold K {0..7,16..23} / {8..15,24..31})
__device__ __forceinline__ bf16x16 frag_a(const __bf16* m, int row0, int kb, int hl, int mr) {
  const __bf16* p = m + (size_t)(row0 + mr) * LDW + kb + hl * 8;
  bf16x8 lo = *(const bf16x8*)p;
  bf16x8 hi = *(const bf16x8*)(p + 16);
  return __builtin_shufflevector(lo, hi, 0,1,2,3,4,5,6,7,8,9,10,11,12,13,14,15);
}
// B-fragment 32x16 bf16 from TRANSPOSED weights Wt[n][k] (lanes 0..15: K {0..7,16..23}, lanes 16..31: K {8..15,24..31})
__device__ __forceinline__ bf16x16 frag_b(const __bf16* wt, int n0, int kb, int hl, int nc) {
  const __bf16* p = wt + (size_t)(n0 + nc) * LDW + kb + hl * 8;
  bf16x8 lo = *(const bf16x8*)p;
  bf16x8 hi = *(const bf16x8*)(p + 16);
  return __builtin_shufflevector(lo, hi, 0,1,2,3,4,5,6,7,8,9,10,11,12,13,14,15);
}
__device__ __forceinline__ f32x8 wmma_bf16(bf16x16 a, bf16x16 b, f32x8 c) {
  return __builtin_amdgcn_wmma_f32_16x16x32_bf16(false, a, false, b, (short)0, c, false, false);
}

// ---------------- workspace layout (floats) ----------------
constexpr size_t WS_MSG1 = 0;                       // [B,N,H]
constexpr size_t WS_MSG2 = WS_MSG1 + 65536;         // [B,N,H]
constexpr size_t WS_O1Z  = WS_MSG2 + 65536;         // [B,N,H]
constexpr size_t WS_TRI1 = WS_O1Z  + 65536;         // [B,N,8]
constexpr size_t WS_TRI2 = WS_TRI1 + 4096;          // [B,N,8]
constexpr size_t WS_MSGG = WS_TRI2 + 4096;          // [B,H]
constexpr size_t WS_TRIG = WS_MSGG + 512;           // [B,8]
constexpr size_t WS_TE1  = 205376;                  // [B,N,N,8]
constexpr size_t WS_TE2  = WS_TE1 + 1048576;
constexpr size_t WS_TE3  = WS_TE2 + 1048576;
constexpr size_t WS_AGG  = WS_TE3 + 1048576;        // [B,N,H]

// =================== kernel A: z-projections (small, VALU) ===================
__global__ __launch_bounds__(256) void pgn_kA(
    const float* __restrict__ node, const float* __restrict__ hidden, const float* __restrict__ graph,
    const float* __restrict__ m1_w, const float* __restrict__ m1_b,
    const float* __restrict__ m2_w, const float* __restrict__ m2_b,
    const float* __restrict__ o1_w, const float* __restrict__ o1_b,
    const float* __restrict__ t1_w, const float* __restrict__ t1_b,
    const float* __restrict__ t2_w, const float* __restrict__ t2_b,
    const float* __restrict__ mg_w, const float* __restrict__ mg_b,
    const float* __restrict__ tg_w, const float* __restrict__ tg_b,
    float* __restrict__ msg1, float* __restrict__ msg2, float* __restrict__ o1z,
    float* __restrict__ tri1, float* __restrict__ tri2,
    float* __restrict__ msgG, float* __restrict__ triG)
{
  __shared__ float z[2 * Fn];
  const int tid = threadIdx.x;
  const int r = blockIdx.x;
  if (r < Bn * Nn) {
    const int b = r >> 7, n = r & 127;
    if (tid < Fn) z[tid] = node[((size_t)b * Nn + n) * Fn + tid];
    else          z[tid] = hidden[((size_t)b * Nn + n) * Fn + (tid - Fn)];
    __syncthreads();
    {
      const float* W; const float* Bv; float* Op; int col;
      if (tid < Hn) { W = m1_w; Bv = m1_b; Op = msg1 + (size_t)r * Hn; col = tid; }
      else          { W = m2_w; Bv = m2_b; Op = msg2 + (size_t)r * Hn; col = tid - Hn; }
      float acc = Bv[col];
      for (int k = 0; k < 2 * Fn; ++k) acc += z[k] * W[k * Hn + col];
      Op[col] = acc;
    }
    if (tid < Hn) {
      float acc = o1_b[tid];
      for (int k = 0; k < 2 * Fn; ++k) acc += z[k] * o1_w[k * Hn + tid];
      o1z[(size_t)r * Hn + tid] = acc;
    } else if (tid < Hn + 8) {
      const int c = tid - Hn;
      float acc = t1_b[c];
      for (int k = 0; k < 2 * Fn; ++k) acc += z[k] * t1_w[k * 8 + c];
      tri1[(size_t)r * 8 + c] = acc;
    } else if (tid < Hn + 16) {
      const int c = tid - Hn - 8;
      float acc = t2_b[c];
      for (int k = 0; k < 2 * Fn; ++k) acc += z[k] * t2_w[k * 8 + c];
      tri2[(size_t)r * 8 + c] = acc;
    }
  } else {
    const int b = r - Bn * Nn;
    if (tid < Fn) z[tid] = graph[(size_t)b * Fn + tid];
    __syncthreads();
    if (tid < Hn) {
      float acc = mg_b[tid];
      for (int k = 0; k < Fn; ++k) acc += z[k] * mg_w[k * Hn + tid];
      msgG[(size_t)b * Hn + tid] = acc;
    } else if (tid < Hn + 8) {
      const int c = tid - Hn;
      float acc = tg_b[c];
      for (int k = 0; k < Fn; ++k) acc += z[k] * tg_w[k * 8 + c];
      triG[(size_t)b * 8 + c] = acc;
    }
  }
}

// ==== kernel B: fused edge projections + msg-MLP + masked max over i (WMMA bf16) ====
// One WG per (b, j). Rows = all 128 i. Everything stays on-WGP; msgs never hits HBM.
__global__ __launch_bounds__(256) void pgn_kB(
    const float* __restrict__ edge, const int* __restrict__ adj,
    const float* __restrict__ me_w, const float* __restrict__ me_b,
    const float* __restrict__ mlp1_w, const float* __restrict__ mlp1_b,
    const float* __restrict__ mlp2_w, const float* __restrict__ mlp2_b,
    const float* __restrict__ te1_w, const float* __restrict__ te1_b,
    const float* __restrict__ te2_w, const float* __restrict__ te2_b,
    const float* __restrict__ te3_w, const float* __restrict__ te3_b,
    const float* __restrict__ msg1, const float* __restrict__ msg2, const float* __restrict__ msgG,
    float* __restrict__ te1o, float* __restrict__ te2o, float* __restrict__ te3o,
    float* __restrict__ agg)
{
  extern __shared__ char smem[];
  __bf16* E   = (__bf16*)smem;            // 128 x LDW : edge tile, later reused in-place as activations
  __bf16* MeT = E   + 128 * LDW;          // transposed weights [n][k]
  __bf16* M1T = MeT + 128 * LDW;
  __bf16* M2T = M1T + 128 * LDW;
  __bf16* TeT = M2T + 128 * LDW;          // 32 x LDW (te1|te2|te3|pad)
  float* bvec = (float*)(TeT + 32 * LDW); // me_b + msg1[b,j,:] + msgG[b,:]
  float* b1v  = bvec + 128;
  float* b2v  = b1v + 128;
  float* tebv = b2v + 128;                // 32
  float* wagg = tebv + 32;                // 8 x 128 per-wave masked maxima

  const int tid = threadIdx.x;
  const int b = blockIdx.x >> 7, j = blockIdx.x & 127;

  for (int idx = tid; idx < 128 * Fn; idx += 256) {           // E[i][k] = edge[b,i,j,k]
    const int i = idx >> 7, k = idx & 127;
    E[i * LDW + k] = (__bf16)edge[(((size_t)b * Nn + i) * Nn + j) * Fn + k];
  }
  for (int idx = tid; idx < 128 * 128; idx += 256) {          // transposed weights
    const int n = idx >> 7, k = idx & 127;
    MeT[n * LDW + k] = (__bf16)me_w[k * Hn + n];
    M1T[n * LDW + k] = (__bf16)mlp1_w[k * Hn + n];
    M2T[n * LDW + k] = (__bf16)mlp2_w[k * Hn + n];
  }
  for (int idx = tid; idx < 32 * 128; idx += 256) {
    const int n = idx >> 7, k = idx & 127;
    float w = 0.f;
    if (n < 8)       w = te1_w[k * 8 + n];
    else if (n < 16) w = te2_w[k * 8 + n - 8];
    else if (n < 24) w = te3_w[k * 8 + n - 16];
    TeT[n * LDW + k] = (__bf16)w;
  }
  if (tid < 128) {
    bvec[tid] = me_b[tid] + msg1[((size_t)b * Nn + j) * Hn + tid] + msgG[(size_t)b * Hn + tid];
    b1v[tid] = mlp1_b[tid];
    b2v[tid] = mlp2_b[tid];
  }
  if (tid < 32) {
    float tb = 0.f;
    if (tid < 8)       tb = te1_b[tid];
    else if (tid < 16) tb = te2_b[tid - 8];
    else if (tid < 24) tb = te3_b[tid - 16];
    tebv[tid] = tb;
  }
  __syncthreads();

  const int lane = tid & 31, wv = tid >> 5;
  const int hl = lane >> 4, nc = lane & 15;
  const int row0 = wv * 16;                                   // this wave's 16 i-rows

  // ---- phase 1: msg_e (8 tiles) + tri_e (2 tiles), K = 128 ----
  f32x8 acc[8] = {}; f32x8 tacc[2] = {};
  for (int kb = 0; kb < 128; kb += 32) {
    const bf16x16 a = frag_a(E, row0, kb, hl, nc);
#pragma unroll
    for (int ht = 0; ht < 8; ++ht)
      acc[ht] = wmma_bf16(a, frag_b(MeT, ht * 16, kb, hl, nc), acc[ht]);
    tacc[0] = wmma_bf16(a, frag_b(TeT, 0,  kb, hl, nc), tacc[0]);
    tacc[1] = wmma_bf16(a, frag_b(TeT, 16, kb, hl, nc), tacc[1]);
  }
  // epilogue: broadcasts + ReLU -> bf16 back into E (wave-local rows, no barrier needed)
#pragma unroll
  for (int ht = 0; ht < 8; ++ht) {
    const int h = ht * 16 + nc;
#pragma unroll
    for (int v = 0; v < 8; ++v) {
      const int i = row0 + hl * 8 + v;
      float val = acc[ht][v] + bvec[h] + msg2[((size_t)b * Nn + i) * Hn + h];
      E[i * LDW + h] = (__bf16)fmaxf(val, 0.f);
    }
  }
#pragma unroll
  for (int v = 0; v < 8; ++v) {                               // triplet edge projections
    const int i = row0 + hl * 8 + v;
    const size_t base = (((size_t)b * Nn + i) * Nn + j) * 8;
    const float t0 = tacc[0][v] + tebv[nc];
    const float t1 = tacc[1][v] + tebv[16 + nc];
    if (nc < 8) te1o[base + nc] = t0; else te2o[base + nc - 8] = t0;
    if (nc < 8) te3o[base + nc] = t1;
  }

  // ---- phase 2: mlp1 + ReLU (in place) ----
  f32x8 acc2[8] = {};
  for (int kb = 0; kb < 128; kb += 32) {
    const bf16x16 a = frag_a(E, row0, kb, hl, nc);
#pragma unroll
    for (int ht = 0; ht < 8; ++ht)
      acc2[ht] = wmma_bf16(a, frag_b(M1T, ht * 16, kb, hl, nc), acc2[ht]);
  }
#pragma unroll
  for (int ht = 0; ht < 8; ++ht) {
    const int h = ht * 16 + nc;
#pragma unroll
    for (int v = 0; v < 8; ++v) {
      const int i = row0 + hl * 8 + v;
      E[i * LDW + h] = (__bf16)fmaxf(acc2[ht][v] + b1v[h], 0.f);
    }
  }

  // ---- phase 3: mlp2 + adjacency-masked max over this wave's i rows ----
  f32x8 acc3[8] = {};
  for (int kb = 0; kb < 128; kb += 32) {
    const bf16x16 a = frag_a(E, row0, kb, hl, nc);
#pragma unroll
    for (int ht = 0; ht < 8; ++ht)
      acc3[ht] = wmma_bf16(a, frag_b(M2T, ht * 16, kb, hl, nc), acc3[ht]);
  }
  int adjv[8];
#pragma unroll
  for (int v = 0; v < 8; ++v)
    adjv[v] = adj[((size_t)b * Nn + (row0 + hl * 8 + v)) * Nn + j];
#pragma unroll
  for (int ht = 0; ht < 8; ++ht) {
    const int h = ht * 16 + nc;
    float local = -1e9f;
#pragma unroll
    for (int v = 0; v < 8; ++v)
      if (adjv[v] == 1) local = fmaxf(local, acc3[ht][v] + b2v[h]);
    local = fmaxf(local, __shfl_xor(local, 16, 32));          // combine the two row halves
    if (hl == 0) wagg[wv * 128 + h] = local;
  }
  __syncthreads();
  if (tid < 128) {                                            // reduce 8 waves -> agg[b,j,:]
    float m = -1e9f;
#pragma unroll
    for (int ww = 0; ww < 8; ++ww) m = fmaxf(m, wagg[ww * 128 + tid]);
    agg[((size_t)b * Nn + j) * Hn + tid] = m;
  }
}

// ==== kernel C: triplet max-plus reduction + o3 GEMV + ReLU + LayerNorm ====
// WG per (b, 16x16 (j,k) tile); thread owns one (j,k); output staged in LDS for coalesced stores.
__global__ __launch_bounds__(256) void pgn_kC(
    const float* __restrict__ tri1, const float* __restrict__ tri2, const float* __restrict__ triG,
    const float* __restrict__ te1o, const float* __restrict__ te2o, const float* __restrict__ te3o,
    const float* __restrict__ o3_w, const float* __restrict__ o3_b,
    const float* __restrict__ scale, const float* __restrict__ bias,
    float* __restrict__ outTri)
{
  extern __shared__ char smem[];
  float* region = (float*)smem;       // 33792 floats
  float* P  = region;                 // [i][jj][t]  (tri1[i]+te1[i,j])
  float* Q  = region + 16384;         // [i][kk][t]  (te2[i,k])
  float* Yt = region;                 // reused: [slot][132] staging (slot = jj*16+kk)
  float* W3 = region + 33792;         // [8][128]
  float* b3 = W3 + 1024;
  float* scl = b3 + 128;
  float* bia = scl + 128;

  const int tid = threadIdx.x;
  const int b = blockIdx.z, j0 = blockIdx.y * 16, k0 = blockIdx.x * 16;

  for (int idx = tid; idx < 16384; idx += 256) {
    const int i = idx >> 7, slot = (idx >> 3) & 15, t = idx & 7;
    P[idx] = tri1[((size_t)b * Nn + i) * 8 + t] +
             te1o[(((size_t)b * Nn + i) * Nn + (j0 + slot)) * 8 + t];
    Q[idx] = te2o[(((size_t)b * Nn + i) * Nn + (k0 + slot)) * 8 + t];
  }
  for (int idx = tid; idx < 1024; idx += 256) W3[idx] = o3_w[idx];
  if (tid < 128) { b3[tid] = o3_b[tid]; scl[tid] = scale[tid]; bia[tid] = bias[tid]; }

  const int jj = tid >> 4, kk = tid & 15;
  const int j = j0 + jj, k = k0 + kk;
  float ct[8];
#pragma unroll
  for (int t = 0; t < 8; ++t)
    ct[t] = tri2[((size_t)b * Nn + j) * 8 + t] + tri1[((size_t)b * Nn + k) * 8 + t]   // tri_3 == tri_1
          + te3o[(((size_t)b * Nn + j) * Nn + k) * 8 + t] + triG[(size_t)b * 8 + t];
  __syncthreads();

  float m[8];
#pragma unroll
  for (int t = 0; t < 8; ++t) m[t] = -1e30f;
  for (int i = 0; i < Nn; ++i) {
    const float* pp = P + ((size_t)i * 16 + jj) * 8;
    const float* qq = Q + ((size_t)i * 16 + kk) * 8;
#pragma unroll
    for (int t = 0; t < 8; ++t) m[t] = fmaxf(m[t], pp[t] + qq[t]);
  }
#pragma unroll
  for (int t = 0; t < 8; ++t) m[t] += ct[t];
  __syncthreads();                                            // P/Q dead -> Yt

  float* yrow = Yt + tid * 132;
  float s1 = 0.f, s2 = 0.f;
  for (int o = 0; o < 128; ++o) {
    float y = b3[o];
#pragma unroll
    for (int t = 0; t < 8; ++t) y += m[t] * W3[t * 128 + o];
    y = fmaxf(y, 0.f);
    yrow[o] = y; s1 += y; s2 += y * y;
  }
  const float mean = s1 * (1.f / 128.f);
  const float var  = s2 * (1.f / 128.f) - mean * mean;
  const float rstd = rsqrtf(var + 1e-5f);
  for (int o = 0; o < 128; ++o)
    yrow[o] = (yrow[o] - mean) * rstd * scl[o] + bia[o];
  __syncthreads();

  for (int idx = tid; idx < 8192; idx += 256) {               // coalesced float4 stores
    const int slot = idx >> 5, o4 = idx & 31;
    const int jj2 = slot >> 4, kk2 = slot & 15;
    const float4 v = *((const float4*)(Yt + slot * 132) + o4);
    *((float4*)(outTri + ((((size_t)b * Nn + j0 + jj2) * Nn + k0 + kk2) * (size_t)Hn)) + o4) = v;
  }
}

// ==== kernel D: out = LN(ReLU(o1z + agg @ o2_w + o2_b)) (WMMA bf16) ====
__global__ __launch_bounds__(256) void pgn_kD(
    const float* __restrict__ agg, const float* __restrict__ o1z,
    const float* __restrict__ o2_w, const float* __restrict__ o2_b,
    const float* __restrict__ scale, const float* __restrict__ bias,
    float* __restrict__ outp)
{
  extern __shared__ char smem[];
  __bf16* A  = (__bf16*)smem;         // 128 x LDW
  __bf16* Wt = A + 128 * LDW;         // transposed o2_w
  float*  Y  = (float*)(Wt + 128 * LDW);  // 128 x 132
  float*  bv = Y + 128 * 132;
  const int tid = threadIdx.x, b = blockIdx.x;

  for (int idx = tid; idx < 128 * 128; idx += 256) {
    const int r = idx >> 7, k = idx & 127;
    A[r * LDW + k]  = (__bf16)agg[((size_t)b * Nn + r) * Hn + k];
    Wt[r * LDW + k] = (__bf16)o2_w[k * Hn + r];
  }
  if (tid < 128) bv[tid] = o2_b[tid];
  __syncthreads();

  const int lane = tid & 31, wv = tid >> 5, hl = lane >> 4, nc = lane & 15;
  const int row0 = wv * 16;
  f32x8 acc[8] = {};
  for (int kb = 0; kb < 128; kb += 32) {
    const bf16x16 a = frag_a(A, row0, kb, hl, nc);
#pragma unroll
    for (int ht = 0; ht < 8; ++ht)
      acc[ht] = wmma_bf16(a, frag_b(Wt, ht * 16, kb, hl, nc), acc[ht]);
  }
#pragma unroll
  for (int ht = 0; ht < 8; ++ht) {
    const int h = ht * 16 + nc;
#pragma unroll
    for (int v = 0; v < 8; ++v) {
      const int i = row0 + hl * 8 + v;
      const float y = acc[ht][v] + bv[h] + o1z[((size_t)b * Nn + i) * Hn + h];
      Y[i * 132 + h] = fmaxf(y, 0.f);
    }
  }
  __syncthreads();
  if (tid < 128) {
    float s1 = 0.f, s2 = 0.f;
    for (int h = 0; h < 128; ++h) { const float v = Y[tid * 132 + h]; s1 += v; s2 += v * v; }
    const float mean = s1 * (1.f / 128.f);
    const float var  = s2 * (1.f / 128.f) - mean * mean;
    const float rstd = rsqrtf(var + 1e-5f);
    for (int h = 0; h < 128; ++h)
      outp[((size_t)b * Nn + tid) * Hn + h] = (Y[tid * 132 + h] - mean) * rstd * scale[h] + bias[h];
  }
}

extern "C" void kernel_launch(void* const* d_in, const int* in_sizes, int n_in,
                              void* d_out, int out_size, void* d_ws, size_t ws_size,
                              hipStream_t stream) {
  const float* node   = (const float*)d_in[0];
  const float* edge   = (const float*)d_in[1];
  const float* graph  = (const float*)d_in[2];
  const float* hidden = (const float*)d_in[3];
  const int*   adj    = (const int*)d_in[4];
  const float* m1_w = (const float*)d_in[5],  *m1_b = (const float*)d_in[6];
  const float* m2_w = (const float*)d_in[7],  *m2_b = (const float*)d_in[8];
  const float* me_w = (const float*)d_in[9],  *me_b = (const float*)d_in[10];
  const float* mg_w = (const float*)d_in[11], *mg_b = (const float*)d_in[12];
  const float* mlp1_w = (const float*)d_in[13], *mlp1_b = (const float*)d_in[14];
  const float* mlp2_w = (const float*)d_in[15], *mlp2_b = (const float*)d_in[16];
  const float* o1_w = (const float*)d_in[17], *o1_b = (const float*)d_in[18];
  const float* o2_w = (const float*)d_in[19], *o2_b = (const float*)d_in[20];
  const float* t1_w = (const float*)d_in[21], *t1_b = (const float*)d_in[22];
  const float* t2_w = (const float*)d_in[23], *t2_b = (const float*)d_in[24];
  const float* te1_w = (const float*)d_in[25], *te1_b = (const float*)d_in[26];
  const float* te2_w = (const float*)d_in[27], *te2_b = (const float*)d_in[28];
  const float* te3_w = (const float*)d_in[29], *te3_b = (const float*)d_in[30];
  const float* tg_w = (const float*)d_in[31], *tg_b = (const float*)d_in[32];
  const float* o3_w = (const float*)d_in[33], *o3_b = (const float*)d_in[34];
  const float* nscale = (const float*)d_in[35], *nbias = (const float*)d_in[36];

  float* ws = (float*)d_ws;
  float* msg1 = ws + WS_MSG1; float* msg2 = ws + WS_MSG2; float* o1z = ws + WS_O1Z;
  float* tri1 = ws + WS_TRI1; float* tri2 = ws + WS_TRI2;
  float* msgG = ws + WS_MSGG; float* triG = ws + WS_TRIG;
  float* te1o = ws + WS_TE1;  float* te2o = ws + WS_TE2;  float* te3o = ws + WS_TE3;
  float* aggp = ws + WS_AGG;

  float* outp   = (float*)d_out;                  // [B,N,H]   = 65536
  float* outTri = outp + (size_t)Bn * Nn * Hn;    // [B,N,N,H] = 8388608

  constexpr size_t SMEM_B = (size_t)(128 * 4 + 32) * LDW * 2 + (size_t)(128 * 3 + 32 + 8 * 128) * 4; // 153728
  constexpr size_t SMEM_C = (size_t)(33792 + 1024 + 384) * 4;                                        // 140800
  constexpr size_t SMEM_D = (size_t)256 * LDW * 2 + (size_t)(128 * 132 + 128) * 4;                   // 137728

  pgn_kA<<<Bn * Nn + Bn, 256, 0, stream>>>(node, hidden, graph,
      m1_w, m1_b, m2_w, m2_b, o1_w, o1_b, t1_w, t1_b, t2_w, t2_b,
      mg_w, mg_b, tg_w, tg_b, msg1, msg2, o1z, tri1, tri2, msgG, triG);

  pgn_kB<<<Bn * Nn, 256, SMEM_B, stream>>>(edge, adj,
      me_w, me_b, mlp1_w, mlp1_b, mlp2_w, mlp2_b,
      te1_w, te1_b, te2_w, te2_b, te3_w, te3_b,
      msg1, msg2, msgG, te1o, te2o, te3o, aggp);

  pgn_kC<<<dim3(Nn / 16, Nn / 16, Bn), 256, SMEM_C, stream>>>(tri1, tri2, triG,
      te1o, te2o, te3o, o3_w, o3_b, nscale, nbias, outTri);

  pgn_kD<<<Bn, 256, SMEM_D, stream>>>(aggp, o1z, o2_w, o2_b, nscale, nbias, outp);
}
